// RNNMemoryModel_35682588295847
// MI455X (gfx1250) — matrix operand order repaired and checked by reading, not compiled, not run on hardware
//
#include <hip/hip_runtime.h>

typedef __attribute__((ext_vector_type(16))) _Float16 v16h;
typedef __attribute__((ext_vector_type(8)))  float    v8f;

namespace {
constexpr int   kN     = 1024;   // neurons
constexpr int   kBatch = 128;
constexpr int   kSeq   = 500;
constexpr int   kInDim = 24;     // 3 * max_item_num
constexpr int   kInPad = 32;     // padded K for the drive WMMA chunk
constexpr float kDt    = 0.1f;
}

union H8 {
  uint4    q;
  _Float16 h[8];
};
union Frag {
  uint4 q[2];
  v16h  h;
};

__device__ __forceinline__ float hw_tanh(float x) {
#if __has_builtin(__builtin_amdgcn_tanhf)
  return __builtin_amdgcn_tanhf(x);
#else
  float y;
  asm("v_tanh_f32 %0, %1" : "=v"(y) : "v"(x));
  return y;
#endif
}

// One-time prep:
//   effW16[n][k] = (f16)(W[n][k] * dales_sign[k])      (1024x1024)
//   Bpad16[n][c] = c<24 ? (f16)B[n][c] : 0             (1024x32)
//   kt[n]        = DT / tau[n]
//   r16_0        = 0  (relu'd f16 state)
__global__ void rnn_prep_kernel(const float* __restrict__ W,
                                const float* __restrict__ dsign,
                                const float* __restrict__ Bin,
                                const float* __restrict__ tau,
                                _Float16* __restrict__ effW16,
                                _Float16* __restrict__ Bpad16,
                                float* __restrict__ kt,
                                _Float16* __restrict__ r16_0) {
  int idx = blockIdx.x * blockDim.x + threadIdx.x;
  if (idx < kN * kN) {
    int k = idx & (kN - 1);
    effW16[idx] = (_Float16)(W[idx] * dsign[k]);
  }
  if (idx < kN * kInPad) {
    int n = idx >> 5;
    int c = idx & (kInPad - 1);
    Bpad16[idx] = (c < kInDim) ? (_Float16)Bin[n * kInDim + c] : (_Float16)0.0f;
  }
  if (idx < kN) kt[idx] = kDt / tau[idx];
  if (idx < kBatch * kN) r16_0[idx] = (_Float16)0.0f;
}

// One recurrent step:
//   pre  = relu16(r) @ effW^T + u_t @ B^T      (WMMA f16 -> f32 accum)
//   act  = 8*(1+tanh(0.4*pre - 3))
//   r'   = r + (act - r) * (DT/tau)            (r_old read from out_seq[t-1])
// Grid: 32 WGs; WG tile = 64(M) x 64(N); 8 waves = 4 M-subtiles x 2 N-halves.
// The WG's full 64x1024 f16 weight slice (contiguous 128 KB) is prestaged in
// LDS once -> single barrier; K loop is global A loads + ds_load_b128 B + v_wmma.
__global__ __launch_bounds__(256, 2)
void rnn_step_kernel(const _Float16* __restrict__ r16_cur,
                     _Float16* __restrict__ r16_next,
                     const _Float16* __restrict__ effW16,
                     const _Float16* __restrict__ Bpad16,
                     const float* __restrict__ u,
                     const float* __restrict__ kt,
                     float* __restrict__ out_seq,
                     float* __restrict__ out_final,
                     int t) {
  __shared__ _Float16 lw[64 * kN];   // 128 KB: full weight slice for this WG
  __shared__ _Float16 lu[64 * kInPad];  // 4 KB: u_t tile, K padded 24 -> 32

  const int tid  = threadIdx.x;
  const int wave = tid >> 5;
  const int lane = tid & 31;
  const int l16  = lane & 15;
  const int hi   = lane >> 4;        // half-wave id (VGPR layout split)

  const int m0base = (blockIdx.x & 1) * 64;    // batch-rows tile base
  const int tileN  = (blockIdx.x >> 1) * 64;   // neuron-cols tile base
  const int mw     = wave & 3;                 // M-subtile within WG
  const int nwv    = wave >> 2;                // N-half within WG
  const int m0     = m0base + mw * 16;
  const int m      = m0 + l16;                 // this lane's A-matrix row

  // ---- prestage the WG's contiguous 128 KB weight slice into LDS ----
  {
    const _Float16* src = effW16 + (size_t)tileN * kN;
#pragma unroll
    for (int i = 0; i < 32; ++i) {
      const int idx = tid + 256 * i;           // 8192 x 16 B total
      *(uint4*)(&lw[idx * 8]) = *(const uint4*)(src + (size_t)idx * 8);
    }
  }

  // ---- stage u_t (padded 24 -> 32, f16) into LDS: 8 f16 per thread ----
  {
    const int nn = tid >> 2;           // 0..63 batch-row within tile
    const int cc = (tid & 3) << 3;     // 0,8,16,24
    H8 pu;
    if (cc < kInDim) {
      const float* up = u + ((size_t)(m0base + nn) * kSeq + t) * kInDim + cc;
      float4 x0 = *(const float4*)(up + 0);
      float4 x1 = *(const float4*)(up + 4);
      pu.h[0] = (_Float16)x0.x; pu.h[1] = (_Float16)x0.y;
      pu.h[2] = (_Float16)x0.z; pu.h[3] = (_Float16)x0.w;
      pu.h[4] = (_Float16)x1.x; pu.h[5] = (_Float16)x1.y;
      pu.h[6] = (_Float16)x1.z; pu.h[7] = (_Float16)x1.w;
    } else {
      pu.q = make_uint4(0u, 0u, 0u, 0u);
    }
    *(uint4*)(&lu[nn * kInPad + cc]) = pu.q;
  }

  __syncthreads();   // the only barrier in the step

  v8f acc[2] = {{0, 0, 0, 0, 0, 0, 0, 0}, {0, 0, 0, 0, 0, 0, 0, 0}};

  // ---- K loop: A from f16 state (global, L2-resident), B from LDS ----
  for (int kb = 0; kb < kN; kb += 32) {
    // A fragment (16x32 f16): lane<16 -> K 0..15, lane>=16 -> K 16..31
    const _Float16* ap = r16_cur + (size_t)m * kN + kb + hi * 16;
    Frag af;
    af.q[0] = *(const uint4*)(ap + 0);
    af.q[1] = *(const uint4*)(ap + 8);

    // B fragments: lane<16 -> col n K 0..15 ; lane>=16 -> col n K 16..31
#pragma unroll
    for (int j = 0; j < 2; ++j) {
      const int nn = nwv * 32 + j * 16 + l16;   // row within LDS slice
      Frag bf;
      const _Float16* lp = &lw[(size_t)nn * kN + kb + hi * 16];
      bf.q[0] = *(const uint4*)(lp + 0);
      bf.q[1] = *(const uint4*)(lp + 8);
      acc[j] = __builtin_amdgcn_wmma_f32_16x16x32_f16(
          false, af.h, false, bf.h, (short)0, acc[j], false, false);
    }
  }

  // ---- drive chunk: u_t (padded, from LDS) x Bpad16 (padded, from global) ----
  {
    const int row = mw * 16 + l16;     // batch-row within the WG's u tile
    Frag au;
    au.q[0] = *(const uint4*)(&lu[row * kInPad + hi * 16 + 0]);
    au.q[1] = *(const uint4*)(&lu[row * kInPad + hi * 16 + 8]);

#pragma unroll
    for (int j = 0; j < 2; ++j) {
      const int n = tileN + nwv * 32 + j * 16 + l16;
      Frag bf;
      const _Float16* bp = Bpad16 + (size_t)n * kInPad + hi * 16;
      bf.q[0] = *(const uint4*)(bp + 0);
      bf.q[1] = *(const uint4*)(bp + 8);
      acc[j] = __builtin_amdgcn_wmma_f32_16x16x32_f16(
          false, au.h, false, bf.h, (short)0, acc[j], false, false);
    }
  }

  // ---- epilogue: activation + leaky-integrator update + streaming store ----
#pragma unroll
  for (int j = 0; j < 2; ++j) {
    const int   n    = tileN + nwv * 32 + j * 16 + l16;
    const float ktau = kt[n];
#pragma unroll
    for (int g = 0; g < 8; ++g) {
      // C/D layout: VGPR g holds M=g (lane<16) or M=g+8 (lane>=16), N=lane%16.
      const int    mm   = m0 + g + hi * 8;
      const float  pre  = acc[j][g];
      const float  act  = 8.0f * (1.0f + hw_tanh(0.4f * pre - 3.0f));
      const size_t sidx = ((size_t)mm * kSeq + t) * kN + n;
      const float  rold = (t == 0) ? 0.0f : out_seq[sidx - kN];  // out_seq[t-1]
      const float  rnew = rold + (act - rold) * ktau;
      out_seq[sidx] = rnew;
      r16_next[(size_t)mm * kN + n] = (_Float16)fmaxf(rnew, 0.0f);
      if (t == kSeq - 1) out_final[(size_t)mm * kN + n] = rnew;
    }
  }
}

extern "C" void kernel_launch(void* const* d_in, const int* in_sizes, int n_in,
                              void* d_out, int out_size, void* d_ws, size_t ws_size,
                              hipStream_t stream) {
  // setup_inputs order: u, W, B, dales_sign, tau
  const float* u     = (const float*)d_in[0];
  const float* W     = (const float*)d_in[1];
  const float* Bin   = (const float*)d_in[2];
  const float* dsign = (const float*)d_in[3];
  const float* tau   = (const float*)d_in[4];

  float* out_seq   = (float*)d_out;
  float* out_final = out_seq + (size_t)kBatch * kSeq * kN;

  // Workspace: effW16 (2 MB) | Bpad16 (64 KB) | kt (4 KB) | r16A | r16B (256 KB each)
  char* ws = (char*)d_ws;
  _Float16* effW16 = (_Float16*)ws;  ws += (size_t)kN * kN * sizeof(_Float16);
  _Float16* Bpad16 = (_Float16*)ws;  ws += (size_t)kN * kInPad * sizeof(_Float16);
  float*    kt     = (float*)ws;     ws += (size_t)kN * sizeof(float);
  _Float16* r16A   = (_Float16*)ws;  ws += (size_t)kBatch * kN * sizeof(_Float16);
  _Float16* r16B   = (_Float16*)ws;

  rnn_prep_kernel<<<(kN * kN + 255) / 256, 256, 0, stream>>>(
      W, dsign, Bin, tau, effW16, Bpad16, kt, r16A);

  for (int t = 0; t < kSeq; ++t) {
    const _Float16* src = (t & 1) ? r16B : r16A;
    _Float16*       dst = (t & 1) ? r16A : r16B;
    rnn_step_kernel<<<32, 256, 0, stream>>>(src, dst, effW16, Bpad16, u, kt,
                                            out_seq, out_final, t);
  }
}
